// HybridModel_62405874811593
// MI455X (gfx1250) — matrix operand order repaired and checked by reading, not compiled, not run on hardware
//
#include <hip/hip_runtime.h>

// ---- CDNA5 wave32 WMMA types ----
typedef __attribute__((ext_vector_type(16))) _Float16 v16h;
typedef __attribute__((ext_vector_type(8)))  _Float16 v8h;
typedef __attribute__((ext_vector_type(2)))  __fp16   v2hb;  // cvt_pkrtz return type
typedef __attribute__((ext_vector_type(8)))  float    v8f;
typedef __attribute__((ext_vector_type(4)))  float    v4f;
typedef __attribute__((ext_vector_type(4)))  unsigned v4u;
typedef __attribute__((ext_vector_type(4)))  int      v4i;
typedef v4i __attribute__((address_space(1))) gv4i;   // global AS int4
typedef v4i __attribute__((address_space(3))) sv4i;   // LDS AS int4

#define K1   784
#define KS1  25      // ceil(784/32) K-steps for GEMM1
#define ROWS_PER_WAVE 16
#define WAVES 8
#define ROWS_PER_BLOCK (ROWS_PER_WAVE * WAVES)   // 128

#define W1S_ELEMS (KS1 * 4 * 32 * 16)   // 51200 f16 (pre-swizzled B-frags of W1)
#define W2S_ELEMS (2 * 32 * 16)         // 1024  f16 (pre-swizzled B-frags of W2)
#define WTOT_ELEMS (W1S_ELEMS + W2S_ELEMS)      // 52224 f16 = 104448 B (contiguous)

// Pack 16 f32 -> one 16x32 f16 A-fragment (8 v_cvt_pk_rtz_f16_f32)
__device__ __forceinline__ v16h make_a_frag(v4f u0, v4f u1, v4f u2, v4f u3) {
    union { v2hb p[8]; v16h v; } u;
    u.p[0] = __builtin_amdgcn_cvt_pkrtz(u0[0], u0[1]);
    u.p[1] = __builtin_amdgcn_cvt_pkrtz(u0[2], u0[3]);
    u.p[2] = __builtin_amdgcn_cvt_pkrtz(u1[0], u1[1]);
    u.p[3] = __builtin_amdgcn_cvt_pkrtz(u1[2], u1[3]);
    u.p[4] = __builtin_amdgcn_cvt_pkrtz(u2[0], u2[1]);
    u.p[5] = __builtin_amdgcn_cvt_pkrtz(u2[2], u2[3]);
    u.p[6] = __builtin_amdgcn_cvt_pkrtz(u3[0], u3[1]);
    u.p[7] = __builtin_amdgcn_cvt_pkrtz(u3[2], u3[3]);
    return u.v;
}

// ---------------------------------------------------------------------------
// Prep: convert W1/W2 (f32) -> f16 B-matrix fragments in exact wave32 layout:
//   B 32x16 f16 frag: lane L holds column n = (L&15), K = k0 + (L>=16?8:0)+(i&7)+(i>=8?16:0),
//   i = 0..15 f16 -> one v16h (32B) per lane, fully coalesced.
// W1 K-dim zero-padded 784 -> 800.  w2s is laid out contiguously after w1s.
// ---------------------------------------------------------------------------
__global__ void prep_swizzle(const float* __restrict__ W1,
                             const float* __restrict__ W2,
                             _Float16* __restrict__ w1s,
                             _Float16* __restrict__ w2s) {
    const int t = blockIdx.x * blockDim.x + threadIdx.x;
    if (t < W1S_ELEMS) {
        // flat = ((ks*4 + nt)*32 + L)*16 + i
        const int i  = t & 15;
        const int L  = (t >> 4) & 31;
        const int nt = (t >> 9) & 3;
        const int ks = t >> 11;
        const int n  = nt * 16 + (L & 15);
        const int k  = ks * 32 + ((L >> 4) << 3) + (i & 7) + ((i >> 3) << 4);   // lanes 16-31: K+8
        w1s[t] = (k < K1) ? (_Float16)W1[n * K1 + k] : (_Float16)0.f;
    } else if (t < WTOT_ELEMS) {
        const int u  = t - W1S_ELEMS;
        const int i  = u & 15;
        const int L  = (u >> 4) & 31;
        const int ks = u >> 9;                           // 0..1 (K=64)
        const int j  = L & 15;
        const int k  = ks * 32 + ((L >> 4) << 3) + (i & 7) + ((i >> 3) << 4);
        w2s[u] = (_Float16)W2[j * 64 + k];
    }
}

// ---------------------------------------------------------------------------
// Fused: async-fill W1/W2 frags into LDS once, then
//        GEMM1(WMMA f16, B from LDS) + bias/ReLU + GEMM2(WMMA f16) +
//        bias/ReLU + angles (W3 rows 1,2 only) + analytic circuit:
//        out = sigmoid(5 * sin(a1) * sin(a2))   [RZ cancels; <Z0>=sin a1 sin a2]
// ---------------------------------------------------------------------------
__global__ __launch_bounds__(256)
void fused_kernel(const float* __restrict__ x,
                  const _Float16* __restrict__ wfr,   // w1s||w2s, 104448 B
                  const float* __restrict__ b1,
                  const float* __restrict__ b2,
                  const float* __restrict__ W3,
                  const float* __restrict__ b3,
                  float* __restrict__ out) {
    __shared__ __align__(16) _Float16 sW[WTOT_ELEMS];       // 104448 B: W1S || W2S
    __shared__ __align__(16) _Float16 sH1[WAVES][16 * 72];  // h1 tile, stride 72
    __shared__ __align__(16) float    sH2[WAVES][16 * 17];  // h2 tile, stride 17

    const int tid  = threadIdx.x;
    const int wv   = tid >> 5;
    const int lane = tid & 31;
    const int col  = lane & 15;
    const int hf   = lane >> 4;                 // half-wave: 0 or 1
    const int m0   = blockIdx.x * ROWS_PER_BLOCK + wv * ROWS_PER_WAVE;

    // ---- stage all weight fragments global -> LDS (one linear async copy) ----
    {
        const int total128 = (WTOT_ELEMS * 2) / 16;          // 6528 x 16B
        const v4u* gsrc = (const v4u*)wfr;
        v4u* ldst = (v4u*)sW;
#if __has_builtin(__builtin_amdgcn_global_load_async_to_lds_b128)
        for (int i = tid; i < total128; i += 256)
            __builtin_amdgcn_global_load_async_to_lds_b128(
                (gv4i*)(gsrc + i), (sv4i*)(ldst + i), 0, 0);
#if __has_builtin(__builtin_amdgcn_s_wait_asynccnt)
        __builtin_amdgcn_s_wait_asynccnt(0);
#else
        asm volatile("s_wait_asynccnt 0" ::: "memory");
#endif
#else
        for (int i = tid; i < total128; i += 256)
            ldst[i] = gsrc[i];
#endif
    }
    __syncthreads();

    const float* rowp = x + (size_t)(m0 + col) * K1;   // A row for this lane
    const v16h*  bls  = (const v16h*)sW;               // B-frags now in LDS

    v8f acc0 = {}; v8f acc1 = {}; v8f acc2 = {}; v8f acc3 = {};

#pragma unroll
    for (int ks = 0; ks < KS1; ++ks) {
        // ---- A-frag 16x32 f16: lanes<16 hold K {0..7,16..23}, lanes>=16 {8..15,24..31}
        const int kA = ks * 32 + (hf << 3);
        v4f u0 = *(const v4f*)(rowp + kA);
        v4f u1 = *(const v4f*)(rowp + kA + 4);
        v4f u2 = {}; v4f u3 = {};
        if (kA + 16 < K1) {                     // constant-folds under full unroll
            u2 = *(const v4f*)(rowp + kA + 16);
            u3 = *(const v4f*)(rowp + kA + 20);
        }
        const v16h a = make_a_frag(u0, u1, u2, u3);

        // ---- 4 N-tiles of W1 B-frags from LDS (ds_load_b128, conflict-free) ----
        const v16h* bp = bls + (size_t)(ks * 4) * 32 + lane;
        acc0 = __builtin_amdgcn_wmma_f32_16x16x32_f16(false, a, false, bp[0],  (short)0, acc0, false, false);
        acc1 = __builtin_amdgcn_wmma_f32_16x16x32_f16(false, a, false, bp[32], (short)0, acc1, false, false);
        acc2 = __builtin_amdgcn_wmma_f32_16x16x32_f16(false, a, false, bp[64], (short)0, acc2, false, false);
        acc3 = __builtin_amdgcn_wmma_f32_16x16x32_f16(false, a, false, bp[96], (short)0, acc3, false, false);
    }

    // ---- bias + ReLU; stash h1 tile (16x64) as f16 in LDS
    // C-frag: VGPR r -> row (hf?8:0)+r, column nt*16+col
    const float bb0 = b1[col], bb1 = b1[col + 16], bb2 = b1[col + 32], bb3 = b1[col + 48];
    const int r0 = hf << 3;
    _Float16* hbase = &sH1[wv][0];
#pragma unroll
    for (int r = 0; r < 8; ++r) {
        const int row = r0 + r;
        hbase[row * 72 + col     ] = (_Float16)fmaxf(acc0[r] + bb0, 0.f);
        hbase[row * 72 + col + 16] = (_Float16)fmaxf(acc1[r] + bb1, 0.f);
        hbase[row * 72 + col + 32] = (_Float16)fmaxf(acc2[r] + bb2, 0.f);
        hbase[row * 72 + col + 48] = (_Float16)fmaxf(acc3[r] + bb3, 0.f);
    }
    __syncthreads();

    // ---- GEMM2: h2(16x16) = h1(16x64) x W2^T via 2 WMMAs (K=64)
    v8f acc = {};
    const _Float16* hp = &sH1[wv][col * 72];    // A row = col
    const v16h* w2f = (const v16h*)(sW + W1S_ELEMS);
#pragma unroll
    for (int ks = 0; ks < 2; ++ks) {
        const int kA = ks * 32 + (hf << 3);
        v8h c0 = *(const v8h*)(hp + kA);
        v8h c1 = *(const v8h*)(hp + kA + 16);
        v16h a;
#pragma unroll
        for (int i = 0; i < 8; ++i) { a[i] = c0[i]; a[8 + i] = c1[i]; }
        acc = __builtin_amdgcn_wmma_f32_16x16x32_f16(false, a, false, w2f[ks * 32 + lane], (short)0, acc, false, false);
    }
    const float b2v = b2[col];
#pragma unroll
    for (int r = 0; r < 8; ++r)
        sH2[wv][(r0 + r) * 17 + col] = fmaxf(acc[r] + b2v, 0.f);
    __syncthreads();

    // ---- tail: angles (W3 rows 1,2 only) + analytic 3-qubit circuit
    if (lane < 16) {
        const int row = lane;
        float d1 = b3[1], d2 = b3[2];
#pragma unroll
        for (int j = 0; j < 16; ++j) {
            const float h = sH2[wv][row * 17 + j];
            d1 = fmaf(h, W3[16 + j], d1);
            d2 = fmaf(h, W3[32 + j], d2);
        }
        const float PI_F = 3.14159265358979f;
        const float a1 = PI_F * tanhf(d1);
        const float a2 = PI_F * tanhf(d2);
        // RZ is diagonal -> cancels in |amp|^2; CNOT chain => final b0 = b1^b2,
        // so <Z0> = (alpha^2-beta^2)_q1 * (alpha^2-beta^2)_q2 = sin(a1)*sin(a2).
        const float z = 5.f * __sinf(a1) * __sinf(a2);
        out[m0 + row] = 1.f / (1.f + __expf(-z));
    }
}

// ---------------------------------------------------------------------------
extern "C" void kernel_launch(void* const* d_in, const int* in_sizes, int n_in,
                              void* d_out, int out_size, void* d_ws, size_t ws_size,
                              hipStream_t stream) {
    const float* x  = (const float*)d_in[0];
    const float* W1 = (const float*)d_in[1];
    const float* b1 = (const float*)d_in[2];
    const float* W2 = (const float*)d_in[3];
    const float* b2 = (const float*)d_in[4];
    const float* W3 = (const float*)d_in[5];
    const float* b3 = (const float*)d_in[6];
    // d_in[7] = q_params: provably unused (RZ phases cancel in probabilities)
    float* out = (float*)d_out;

    _Float16* w1s = (_Float16*)d_ws;            // 102400 B
    _Float16* w2s = w1s + W1S_ELEMS;            // +2048 B contiguous (ws >= ~112KB)

    const int B = in_sizes[0] / K1;             // 65536
    prep_swizzle<<<(WTOT_ELEMS + 255) / 256, 256, 0, stream>>>(W1, W2, w1s, w2s);
    fused_kernel<<<B / ROWS_PER_BLOCK, 256, 0, stream>>>(x, w1s, b1, b2, W3, b3, out);
}
